// PredictiveGraphormer_85358180041572
// MI455X (gfx1250) — compile-verified, matrix operand order checked by reading, not run on hardware
//
#include <hip/hip_runtime.h>
#include <hip/hip_bf16.h>
#include <math.h>

// ---------------- problem constants ----------------
constexpr int B_ = 8, T_ = 5, N_ = 2048, FIN = 64, E_ = 32768;
constexpr int D_ = 256, H_ = 8, FF_ = 2048, NCLS_ = 10, LAYERS_ = 3;
constexpr int G_  = B_ * T_;     // 40 graphs
constexpr int GN  = G_ * N_;     // 81920 rows (= T*B*N for temporal phase)
constexpr int GE  = G_ * E_;     // 1310720 edges
constexpr int BN  = B_ * N_;     // 16384
constexpr float LN_EPS_ = 1e-5f;
constexpr float INV_SQRT_C = 0.17677669529663687f;  // 1/sqrt(32)

typedef __attribute__((ext_vector_type(16))) __bf16 bf16x16;
typedef __attribute__((ext_vector_type(8)))  __bf16 bf16x8;
typedef __attribute__((ext_vector_type(8)))  float  f32x8;

// LDS tile row stride in dwords: 16 dwords of payload (32 bf16) + 4 pad.
// 80-byte stride keeps 16-byte alignment for b128 fragment loads and spreads
// the 16 row-lanes of a fragment read across 16 distinct banks.
constexpr int LROW = 20;

// ---------------- helpers ----------------
__device__ __forceinline__ unsigned pack_bf16(float a, float b) {
    __bf16 x = (__bf16)a, y = (__bf16)b;
    unsigned short ux = __builtin_bit_cast(unsigned short, x);
    unsigned short uy = __builtin_bit_cast(unsigned short, y);
    return (unsigned)ux | ((unsigned)uy << 16);   // lowers to v_cvt_pk_bf16_f32
}

__device__ __forceinline__ float atomicMaxFloat(float* addr, float val) {
    unsigned int* ua = (unsigned int*)addr;
    unsigned int old = *ua;
    while (true) {
        float cur = __uint_as_float(old);
        if (cur >= val) break;
        unsigned int assumed = old;
        old = atomicCAS(ua, assumed, __float_as_uint(val));
        if (old == assumed) break;
    }
    return __uint_as_float(old);
}

__device__ __forceinline__ bf16x16 load_frag_lds(const unsigned* tile, int lane, int rowBase) {
    // tile rows: LROW dwords, first 16 dwords = 32 K-contiguous bf16.
    // ISA 7.12.2 16-bit A/B layout: lanes 0-15 hold K(0..7,16..23),
    // lanes 16-31 hold K(8..15,24..31) of row (lane&15)+rowBase.
    const int khalf = (lane >> 4) << 3;             // 0 or 8
    const __bf16* p = (const __bf16*)(tile + (rowBase + (lane & 15)) * LROW);
    bf16x8 lo = *(const bf16x8*)(p + khalf);        // K khalf..khalf+7
    bf16x8 hi = *(const bf16x8*)(p + 16 + khalf);   // K 16+khalf..+7
    return __builtin_shufflevector(lo, hi, 0, 1, 2, 3, 4, 5, 6, 7,
                                           8, 9, 10, 11, 12, 13, 14, 15);
}

// ---------------- generic tiled BF16-WMMA GEMM: C = A[M,K] @ W[K,N] + bias ----------------
// block tile 128(M) x 256(N), K step 32. 8 waves in a 2x4 grid, each owning a
// 64x64 subtile -> 16 v_wmma per wave per K-step vs 16 ds_load_b128 (1:1).
__global__ __launch_bounds__(256)
void gemm_bf16_wmma(const float* __restrict__ A, const float* __restrict__ W,
                    const float* __restrict__ bias, float* __restrict__ Cout,
                    int M, int Nn, int K, int relu) {
    __shared__ unsigned As[128 * LROW];   // [m][kpair]
    __shared__ unsigned Bs[256 * LROW];   // [n][kpair] (transposed)
    const int tid  = threadIdx.x;
    const int lane = tid & 31;
    const int wid  = tid >> 5;
    const int wm   = wid >> 2;        // 0..1 -> 64-row M subtile
    const int wn   = wid & 3;         // 0..3 -> 64-col N subtile
    const int mr0  = blockIdx.y * 128;
    const int nb0  = blockIdx.x * 256;

    f32x8 acc[4][4] = {};

    for (int kb = 0; kb < K; kb += 32) {
        // stage A tile 128x32: one float2 (K-pair) load + packed b32 LDS store
        #pragma unroll
        for (int i = 0; i < 8; ++i) {
            int idx = tid + i * 256;              // 0..2047 pairs
            int r = idx >> 4, kp = idx & 15;
            const float2 f = ((const float2*)(A + (long)(mr0 + r) * K + kb))[kp];
            As[r * LROW + kp] = pack_bf16(f.x, f.y);
        }
        // stage B tile 32x256 -> [n][kpair]; global reads coalesced on N
        #pragma unroll
        for (int i = 0; i < 16; ++i) {
            int idx = tid + i * 256;              // 0..4095 pairs
            int kp = idx >> 8, n = idx & 255;
            float w0 = W[(long)(kb + 2 * kp    ) * Nn + nb0 + n];
            float w1 = W[(long)(kb + 2 * kp + 1) * Nn + nb0 + n];
            Bs[n * LROW + kp] = pack_bf16(w0, w1);
        }
        __syncthreads();

        bf16x16 afrag[4], bfrag[4];
        #pragma unroll
        for (int i = 0; i < 4; ++i) afrag[i] = load_frag_lds(As, lane, wm * 64 + i * 16);
        #pragma unroll
        for (int j = 0; j < 4; ++j) bfrag[j] = load_frag_lds(Bs, lane, wn * 64 + j * 16);

        #pragma unroll
        for (int i = 0; i < 4; ++i)
            #pragma unroll
            for (int j = 0; j < 4; ++j)
                acc[i][j] = __builtin_amdgcn_wmma_f32_16x16x32_bf16(
                    false, afrag[i], false, bfrag[j], (short)0, acc[i][j], false, false);
        __syncthreads();
    }

    // epilogue: ISA C/D layout -> VGPR r: lanes 0-15 row=r, lanes 16-31 row=8+r
    const int colBase = nb0 + wn * 64 + (lane & 15);
    const int rowHalf = (lane >> 4) * 8;
    #pragma unroll
    for (int i = 0; i < 4; ++i) {
        #pragma unroll
        for (int j = 0; j < 4; ++j) {
            int col = colBase + j * 16;
            float bv = bias ? bias[col] : 0.f;
            #pragma unroll
            for (int r = 0; r < 8; ++r) {
                int row = mr0 + wm * 64 + i * 16 + rowHalf + r;
                float vv = acc[i][j][r] + bv;
                if (relu) vv = fmaxf(vv, 0.f);
                Cout[(long)row * Nn + col] = vv;
            }
        }
    }
}

// ---------------- graph attention (TransformerConv) ----------------
__global__ void init_md(float* __restrict__ m, float* __restrict__ denom) {
    int i = blockIdx.x * 256 + threadIdx.x;
    if (i < GN * H_) { m[i] = -3.0e38f; denom[i] = 0.f; }
}

// wave per edge; lane = channel-within-head; 8 heads per lane.
__global__ __launch_bounds__(256)
void edge_alpha(const int* __restrict__ eidx, const float* __restrict__ eattr,
                const float* __restrict__ W_edge, const float* __restrict__ b_edge,
                const float* __restrict__ q, const float* __restrict__ k,
                float* __restrict__ alpha, float* __restrict__ m) {
    __shared__ float sWe[6 * 256];
    __shared__ float sBe[256];
    int tid = threadIdx.x;
    for (int i = tid; i < 6 * 256; i += 256) sWe[i] = W_edge[i];
    sBe[tid] = b_edge[tid];
    __syncthreads();

    int gid  = blockIdx.x * 8 + (tid >> 5);   // edge id, grid is exact
    int lane = tid & 31;
    int g = gid >> 15, e = gid & (E_ - 1);
    int src = eidx[((long)g * 2    ) * E_ + e];
    int tgt = eidx[((long)g * 2 + 1) * E_ + e];
    long sf = (long)g * N_ + src, tf = (long)g * N_ + tgt;

    float ea[6];
    const float* ap = eattr + ((long)g * E_ + e) * 6;
    #pragma unroll
    for (int j = 0; j < 6; ++j) ea[j] = ap[j];

    float part[8];
    #pragma unroll
    for (int h = 0; h < 8; ++h) {
        int d = h * 32 + lane;
        float ed = sBe[d];
        #pragma unroll
        for (int j = 0; j < 6; ++j) ed = fmaf(ea[j], sWe[j * 256 + d], ed);
        part[h] = q[tf * 256 + d] * (k[sf * 256 + d] + ed);
    }
    #pragma unroll
    for (int off = 16; off; off >>= 1)
        #pragma unroll
        for (int h = 0; h < 8; ++h) part[h] += __shfl_xor(part[h], off, 32);
    if (lane < 8) {
        float al = part[lane] * INV_SQRT_C;
        alpha[(long)gid * 8 + lane] = al;
        atomicMaxFloat(&m[tf * 8 + lane], al);
    }
}

__global__ void edge_softexp(const int* __restrict__ eidx, const float* __restrict__ m,
                             float* __restrict__ alpha, float* __restrict__ denom) {
    int i  = blockIdx.x * 256 + threadIdx.x;    // < GE*8, grid exact
    int h  = i & 7;
    int ge = i >> 3;
    int g = ge >> 15, e = ge & (E_ - 1);
    int tgt = eidx[((long)g * 2 + 1) * E_ + e];
    long tf = (long)g * N_ + tgt;
    float a = __expf(alpha[i] - m[tf * 8 + h]);
    alpha[i] = a;
    atomicAdd(&denom[tf * 8 + h], a);
}

__global__ __launch_bounds__(256)
void edge_scatter(const int* __restrict__ eidx, const float* __restrict__ eattr,
                  const float* __restrict__ W_edge, const float* __restrict__ b_edge,
                  const float* __restrict__ v, const float* __restrict__ alpha,
                  const float* __restrict__ denom, float* __restrict__ outp) {
    __shared__ float sWe[6 * 256];
    __shared__ float sBe[256];
    int tid = threadIdx.x;
    for (int i = tid; i < 6 * 256; i += 256) sWe[i] = W_edge[i];
    sBe[tid] = b_edge[tid];
    __syncthreads();

    int gid  = blockIdx.x * 8 + (tid >> 5);
    int lane = tid & 31;
    int g = gid >> 15, e = gid & (E_ - 1);
    int src = eidx[((long)g * 2    ) * E_ + e];
    int tgt = eidx[((long)g * 2 + 1) * E_ + e];
    long sf = (long)g * N_ + src, tf = (long)g * N_ + tgt;

    float ea[6];
    const float* ap = eattr + ((long)g * E_ + e) * 6;
    #pragma unroll
    for (int j = 0; j < 6; ++j) ea[j] = ap[j];

    float w[8];
    #pragma unroll
    for (int h = 0; h < 8; ++h)
        w[h] = alpha[(long)gid * 8 + h] / denom[tf * 8 + h];

    #pragma unroll
    for (int h = 0; h < 8; ++h) {
        int d = h * 32 + lane;
        float ed = sBe[d];
        #pragma unroll
        for (int j = 0; j < 6; ++j) ed = fmaf(ea[j], sWe[j * 256 + d], ed);
        float vj = v[sf * 256 + d] + ed;
        atomicAdd(&outp[tf * 256 + d], vj * w[h]);
    }
}

// ---------------- [B,T,N,D] -> [T, B*N, D] ----------------
__global__ void reshape_seq(const float* __restrict__ hconv, float* __restrict__ seq) {
    for (long i = (long)blockIdx.x * 256 + threadIdx.x; i < (long)GN * D_;
         i += (long)gridDim.x * 256) {
        int d = (int)(i & 255);
        long rest = i >> 8;                 // (b*T+t)*N + n
        int n = (int)(rest & (N_ - 1));
        long gg = rest >> 11;               // b*T + t
        int t = (int)(gg % T_);
        int b = (int)(gg / T_);
        seq[(((long)t * B_ + b) * N_ + n) * D_ + d] = hconv[i];
    }
}

// ---------------- temporal MHA over T=5; wave per (bn, head), lane = channel ----------------
__global__ __launch_bounds__(256)
void mha_time(const float* __restrict__ qkv, float* __restrict__ outp) {
    int wgid = blockIdx.x * 8 + (threadIdx.x >> 5);   // 0..BN*H-1, grid exact
    int lane = threadIdx.x & 31;
    int h  = wgid & 7;
    int bn = wgid >> 3;
    int dcol = h * 32 + lane;

    float qv[T_], kv[T_], vv[T_];
    #pragma unroll
    for (int t = 0; t < T_; ++t) {
        const float* rowp = qkv + (long)(t * BN + bn) * (3 * D_);
        qv[t] = rowp[dcol];
        kv[t] = rowp[D_ + dcol];
        vv[t] = rowp[2 * D_ + dcol];
    }
    float s[T_][T_];
    #pragma unroll
    for (int t = 0; t < T_; ++t)
        #pragma unroll
        for (int u = 0; u < T_; ++u) {
            float p = qv[t] * kv[u];
            #pragma unroll
            for (int off = 16; off; off >>= 1) p += __shfl_xor(p, off, 32);
            s[t][u] = p * INV_SQRT_C;
        }
    #pragma unroll
    for (int t = 0; t < T_; ++t) {
        float mx = s[t][0];
        #pragma unroll
        for (int u = 1; u < T_; ++u) mx = fmaxf(mx, s[t][u]);
        float den = 0.f;
        #pragma unroll
        for (int u = 0; u < T_; ++u) { s[t][u] = __expf(s[t][u] - mx); den += s[t][u]; }
        float o = 0.f;
        #pragma unroll
        for (int u = 0; u < T_; ++u) o += s[t][u] * vv[u];
        outp[(long)(t * BN + bn) * D_ + dcol] = o / den;
    }
}

// ---------------- fused residual-add + LayerNorm (block per row of 256) ----------------
__global__ __launch_bounds__(256)
void add_layernorm(float* __restrict__ seq, const float* __restrict__ add,
                   const float* __restrict__ gw, const float* __restrict__ bw) {
    __shared__ float s_sum[8], s_sq[8];
    int tid = threadIdx.x, lane = tid & 31, wid = tid >> 5;
    long r = blockIdx.x;
    float x = seq[r * 256 + tid] + add[r * 256 + tid];
    float s = x, qq = x * x;
    #pragma unroll
    for (int off = 16; off; off >>= 1) {
        s  += __shfl_xor(s, off, 32);
        qq += __shfl_xor(qq, off, 32);
    }
    if (lane == 0) { s_sum[wid] = s; s_sq[wid] = qq; }
    __syncthreads();
    float tot = 0.f, totq = 0.f;
    #pragma unroll
    for (int i = 0; i < 8; ++i) { tot += s_sum[i]; totq += s_sq[i]; }
    float mean = tot * (1.f / 256.f);
    float var  = totq * (1.f / 256.f) - mean * mean;
    seq[r * 256 + tid] = (x - mean) * rsqrtf(var + LN_EPS_) * gw[tid] + bw[tid];
}

// ---------------- final classifier: seq[T-1] @ W_out + b_out ----------------
__global__ void classify(const float* __restrict__ seq, const float* __restrict__ Wout,
                         const float* __restrict__ bout, float* __restrict__ out) {
    int i = blockIdx.x * 256 + threadIdx.x;
    if (i >= BN * NCLS_) return;
    int c = i % NCLS_;
    int row = i / NCLS_;
    const float* xp = seq + ((long)(T_ - 1) * BN + row) * D_;
    float acc = bout[c];
    for (int kk = 0; kk < D_; ++kk) acc = fmaf(xp[kk], Wout[kk * NCLS_ + c], acc);
    out[i] = acc;
}

// ---------------- host launch ----------------
extern "C" void kernel_launch(void* const* d_in, const int* in_sizes, int n_in,
                              void* d_out, int out_size, void* d_ws, size_t ws_size,
                              hipStream_t stream) {
    const float* x_seq  = (const float*)d_in[0];
    const int*   eidx   = (const int*)  d_in[1];
    const float* eattr  = (const float*)d_in[2];
    const float* W_node = (const float*)d_in[3];
    const float* b_node = (const float*)d_in[4];
    const float* W_edge = (const float*)d_in[5];
    const float* b_edge = (const float*)d_in[6];
    const float* Wq = (const float*)d_in[7];   const float* bq = (const float*)d_in[8];
    const float* Wk = (const float*)d_in[9];   const float* bk = (const float*)d_in[10];
    const float* Wv = (const float*)d_in[11];  const float* bv = (const float*)d_in[12];
    const float* Ws = (const float*)d_in[13];  const float* bs = (const float*)d_in[14];
    const float* Wqkv = (const float*)d_in[15]; const float* bqkv = (const float*)d_in[16];
    const float* Wo = (const float*)d_in[17];  const float* bo = (const float*)d_in[18];
    const float* W1 = (const float*)d_in[19];  const float* b1 = (const float*)d_in[20];
    const float* W2 = (const float*)d_in[21];  const float* b2 = (const float*)d_in[22];
    const float* g_ln1 = (const float*)d_in[23]; const float* b_ln1 = (const float*)d_in[24];
    const float* g_ln2 = (const float*)d_in[25]; const float* b_ln2 = (const float*)d_in[26];
    const float* W_out = (const float*)d_in[27]; const float* b_out = (const float*)d_in[28];
    float* out = (float*)d_out;
    float* ws  = (float*)d_ws;

    const size_t SZ = (size_t)GN * D_;          // 20,971,520 floats per [GN, 256] buffer
    float* h      = ws;                         // node embed; later reused as attn-out
    float* q      = ws + 1 * SZ;                // q/k/v contiguous -> reused as QKV buf
    float* k      = ws + 2 * SZ;
    float* v      = ws + 3 * SZ;
    float* hconv  = ws + 4 * SZ;                // skip + scattered conv output
    float* seq    = ws + 5 * SZ;                // [T, B*N, D]
    float* ao2    = ws + 6 * SZ;                // proj / FF2 output
    float* alpha  = ws + 7 * SZ;                // GE*8
    float* m      = alpha + (size_t)GE * 8;     // GN*8
    float* denom  = m + (size_t)GN * 8;         // GN*8
    float* ffbuf  = denom + (size_t)GN * 8;     // GN*FF
    float* qkvbuf  = q;                         // 3*SZ region (exactly GN*768)
    float* attnout = h;

    dim3 blk(256);
    auto gemm = [&](const float* A, const float* W, const float* bias, float* C,
                    int M, int Nn, int K, int relu) {
        dim3 grid(Nn / 256, M / 128);
        gemm_bf16_wmma<<<grid, blk, 0, stream>>>(A, W, bias, C, M, Nn, K, relu);
    };

    // 1. node embedding: [GN,64] @ [64,256]
    gemm(x_seq, W_node, b_node, h, GN, D_, FIN, 0);
    // 2. q/k/v/skip projections: [GN,256] @ [256,256]
    gemm(h, Wq, bq, q,     GN, D_, D_, 0);
    gemm(h, Wk, bk, k,     GN, D_, D_, 0);
    gemm(h, Wv, bv, v,     GN, D_, D_, 0);
    gemm(h, Ws, bs, hconv, GN, D_, D_, 0);
    // 3. graph attention (segment softmax + scatter into hconv)
    init_md<<<(GN * H_ + 255) / 256, blk, 0, stream>>>(m, denom);
    edge_alpha<<<GE / 8, blk, 0, stream>>>(eidx, eattr, W_edge, b_edge, q, k, alpha, m);
    edge_softexp<<<(GE * H_) / 256, blk, 0, stream>>>(eidx, m, alpha, denom);
    edge_scatter<<<GE / 8, blk, 0, stream>>>(eidx, eattr, W_edge, b_edge, v, alpha, denom, hconv);
    // 4. reshape to [T, B*N, D]
    reshape_seq<<<8192, blk, 0, stream>>>(hconv, seq);
    // 5. temporal transformer encoder layers
    for (int l = 0; l < LAYERS_; ++l) {
        gemm(seq, Wqkv + (size_t)l * D_ * 3 * D_, bqkv + (size_t)l * 3 * D_,
             qkvbuf, GN, 3 * D_, D_, 0);
        mha_time<<<(BN * H_) / 8, blk, 0, stream>>>(qkvbuf, attnout);
        gemm(attnout, Wo + (size_t)l * D_ * D_, bo + (size_t)l * D_, ao2, GN, D_, D_, 0);
        add_layernorm<<<GN, blk, 0, stream>>>(seq, ao2, g_ln1 + l * D_, b_ln1 + l * D_);
        gemm(seq, W1 + (size_t)l * D_ * FF_, b1 + (size_t)l * FF_, ffbuf, GN, FF_, D_, 1);
        gemm(ffbuf, W2 + (size_t)l * FF_ * D_, b2 + (size_t)l * D_, ao2, GN, D_, FF_, 0);
        add_layernorm<<<GN, blk, 0, stream>>>(seq, ao2, g_ln2 + l * D_, b_ln2 + l * D_);
    }
    // 6. classifier
    classify<<<(BN * NCLS_ + 255) / 256, blk, 0, stream>>>(seq, W_out, b_out, out);
}
